// Lorenz96_correction_7421703487572
// MI455X (gfx1250) — matrix operand order repaired and checked by reading, not compiled, not run on hardware
//
#include <hip/hip_runtime.h>

typedef __attribute__((ext_vector_type(16))) _Float16 v16h;
typedef __attribute__((ext_vector_type(8)))  float    v8f;

#define NB 65536
#define XD 40
#define WAVES_PER_BLOCK 8
#define BLOCK_THREADS 256

// g_pad per wave: 48 channels x 44 (circularly padded positions) = 2112 entries,
// plus a zeroed tail [2112, 2304) so padded K-rows (240..255) read exact zeros.
#define GPAD_MAIN 2112
#define GPAD_SIZE 2304

// im2col row r -> base offset in g_pad (compile-time when r is a constant)
#define OFFC(r) ((((r) / 5) * 44) + ((r) % 5))

union FragAB {
    unsigned short s[16];
    unsigned int   u[8];
    v16h           h;
};

__device__ __forceinline__ unsigned short f2h_bits(float x) {
    _Float16 h = (_Float16)x;
    return __builtin_bit_cast(unsigned short, h);
}

__global__ __launch_bounds__(BLOCK_THREADS)
__attribute__((amdgpu_waves_per_eu(4)))      // cap VGPRs at 256: no MSB mode, >=4 waves/SIMD
void lorenz96_corr_kernel(const float* __restrict__ u,
                          const float* __restrict__ coeff,
                          const float* __restrict__ W1,
                          const float* __restrict__ b1,
                          const float* __restrict__ W2,
                          const float* __restrict__ b2,
                          const float* __restrict__ W3,
                          const float* __restrict__ b3,
                          float* __restrict__ out)
{
    // W2 im2col: M=48 (37 padded), K=256 (240 padded), f16, shared by all waves.
    __shared__ __align__(16) unsigned short A_lds[48 * 256];
    __shared__ __align__(8)  float2         bw_lds[48];   // (b2, W3) per output channel
    __shared__ __align__(16) float          u_pad_all[WAVES_PER_BLOCK][44];
    __shared__ __align__(16) unsigned short g_pad_all[WAVES_PER_BLOCK][GPAD_SIZE];

    const int tid = threadIdx.x;

    // ---- Stage A (W2 im2col) once per block ----
    for (int idx = tid; idx < 48 * 256; idx += BLOCK_THREADS) {
        int m = idx >> 8;
        int r = idx & 255;
        float w = 0.0f;
        if (m < 37 && r < 240) {
            int cin = r / 5;
            int kk  = r - cin * 5;
            w = W2[(m * 48 + cin) * 5 + kk];
        }
        A_lds[idx] = f2h_bits(w);
    }
    if (tid < 48) {
        float bb = (tid < 37) ? b2[tid] : 0.0f;
        float ww = (tid < 37) ? W3[tid] : 0.0f;
        bw_lds[tid] = make_float2(bb, ww);
    }
    // Zero the g_pad tails (read by padded K-rows; must be 0.0, never written later)
    for (int idx = tid; idx < WAVES_PER_BLOCK * (GPAD_SIZE - GPAD_MAIN); idx += BLOCK_THREADS) {
        int w = idx / (GPAD_SIZE - GPAD_MAIN);
        int e = idx - w * (GPAD_SIZE - GPAD_MAIN);
        g_pad_all[w][GPAD_MAIN + e] = 0;
    }
    __syncthreads();

    // Uniform scalars -> SGPRs
    float cf[18];
#pragma unroll
    for (int j = 0; j < 18; ++j) cf[j] = coeff[j];
    const float b3v = b3[0];

    const int wv    = tid >> 5;
    const int lane  = tid & 31;
    const int laneN = lane & 15;
    const int hi    = lane >> 4;          // 0: lanes 0-15, 1: lanes 16-31
    float*          u_pad = u_pad_all[wv];
    unsigned short* g_pad = g_pad_all[wv];

    const int gwave  = blockIdx.x * WAVES_PER_BLOCK + wv;
    const int nwaves = gridDim.x * WAVES_PER_BLOCK;

    for (int n = gwave; n < NB; n += nwaves) {
        const float* urow = u + n * XD;

        // ---- circularly padded u: u_pad[j] = u[(j-2) mod 40], j=0..43 ----
        for (int i = lane; i < 44; i += 32)
            u_pad[i] = urow[(i + 38) % 40];

        // ---- conv1 (f32 VALU): pass-through channels 0..23 (uniform, branch-free) ----
        for (int i = lane; i < 24 * 40; i += 32) {      // 30 full iterations
            int c = i / 40;
            int p = i - c * 40;
            float a = b1[c];
#pragma unroll
            for (int k = 0; k < 5; ++k) a = fmaf(W1[c * 5 + k], u_pad[p + k], a);
            g_pad[c * 44 + p + 2] = f2h_bits(fmaxf(a, 0.0f));
        }
        // ---- gated channels 24..47 = relu(conv(c)) * relu(conv(c+24)) ----
        for (int i = lane; i < 24 * 40; i += 32) {      // 30 full iterations
            int ca = i / 40;
            int p  = i - ca * 40;
            int c  = 24 + ca;
            int c2 = 48 + ca;
            float a = b1[c];
            float b = b1[c2];
#pragma unroll
            for (int k = 0; k < 5; ++k) {
                float uv = u_pad[p + k];
                a = fmaf(W1[c * 5 + k],  uv, a);
                b = fmaf(W1[c2 * 5 + k], uv, b);
            }
            float g = fmaxf(a, 0.0f) * fmaxf(b, 0.0f);
            g_pad[c * 44 + p + 2] = f2h_bits(g);
        }
        // ---- circular halo, branch-free LDS->LDS copy (6 uniform iterations) ----
        for (int t = lane; t < 48 * 4; t += 32) {
            int c = t >> 2;
            int e = t & 3;
            int dst = c * 44 + ((e < 2) ? e : (40 + e));
            int src = c * 44 + ((e < 2) ? (40 + e) : e);
            g_pad[dst] = g_pad[src];
        }

        // ---- conv2 via WMMA: D(48x48) = W2(48x256) x im2col(256x48) ----
        for (int nt = 0; nt < 3; ++nt) {
            const int pcol = nt * 16 + laneN;           // output position (column)

            // B fragments for all 8 K-tiles. One divergent branch on the lane half
            // makes every im2col offset a compile-time constant -> each element is a
            // single ds_load_u16 with an immediate offset off one shared address.
            // K-rows >= 240 hit the zeroed tail; pad columns (pcol>=40) read junk
            // but only feed discarded D columns.
            FragAB bfr[8];
            if (hi == 0) {
#pragma unroll
                for (int kt = 0; kt < 8; ++kt)
#pragma unroll
                    for (int j = 0; j < 8; ++j) {
                        bfr[kt].s[2 * j]     = g_pad[OFFC(kt * 32 + 2 * j)     + pcol];
                        bfr[kt].s[2 * j + 1] = g_pad[OFFC(kt * 32 + 2 * j + 1) + pcol];
                    }
            } else {
#pragma unroll
                for (int kt = 0; kt < 8; ++kt)
#pragma unroll
                    for (int j = 0; j < 8; ++j) {
                        bfr[kt].s[2 * j]     = g_pad[OFFC(kt * 32 + 16 + 2 * j)     + pcol];
                        bfr[kt].s[2 * j + 1] = g_pad[OFFC(kt * 32 + 16 + 2 * j + 1) + pcol];
                    }
            }

            // Opaque offsets: keep A-fragment loads in LDS (no hoist-to-register blowup)
            unsigned aoff0 = (unsigned)(( 0 + laneN) * 256);
            unsigned aoff1 = (unsigned)((16 + laneN) * 256);
            unsigned aoff2 = (unsigned)((32 + laneN) * 256);
            asm volatile("" : "+v"(aoff0), "+v"(aoff1), "+v"(aoff2));

            // Three independent accumulator chains (M-tiles 0..2), C starts at inline 0.
            v8f acc0 = {}, acc1 = {}, acc2 = {};
#pragma unroll
            for (int kt = 0; kt < 8; ++kt) {
                FragAB af0, af1, af2;
#pragma unroll
                for (int j = 0; j < 8; ++j) {
                    int Kl = 2 * j;
                    int K  = kt * 32 + Kl + hi * 16;
                    af0.u[j] = *(const unsigned int*)&A_lds[aoff0 + K];
                    af1.u[j] = *(const unsigned int*)&A_lds[aoff1 + K];
                    af2.u[j] = *(const unsigned int*)&A_lds[aoff2 + K];
                }
                acc0 = __builtin_amdgcn_wmma_f32_16x16x32_f16(
                    false, af0.h, false, bfr[kt].h, (short)0, acc0, false, false);
                acc1 = __builtin_amdgcn_wmma_f32_16x16x32_f16(
                    false, af1.h, false, bfr[kt].h, (short)0, acc1, false, false);
                acc2 = __builtin_amdgcn_wmma_f32_16x16x32_f16(
                    false, af2.h, false, bfr[kt].h, (short)0, acc2, false, false);
            }

            // Epilogue: bias + ReLU + 1x1 conv3 fold; (b2,W3) re-read from LDS
            // behind an opaque offset so they never become spilled live registers.
            unsigned bwoff = (unsigned)(hi * 8);
            asm volatile("" : "+v"(bwoff));
            float partial = 0.0f;
#pragma unroll
            for (int j = 0; j < 8; ++j) {
                float2 bw0 = bw_lds[bwoff +  0 + j];
                float2 bw1 = bw_lds[bwoff + 16 + j];
                float2 bw2 = bw_lds[bwoff + 32 + j];
                partial = fmaf(bw0.y, fmaxf(acc0[j] + bw0.x, 0.0f), partial);
                partial = fmaf(bw1.y, fmaxf(acc1[j] + bw1.x, 0.0f), partial);
                partial = fmaf(bw2.y, fmaxf(acc2[j] + bw2.x, 0.0f), partial);
            }

            // combine the two M-halves (lanes L and L^16 share a column)
            float other = __shfl_xor(partial, 16, 32);

            if (hi == 0 && pcol < XD) {
                float um2 = u_pad[pcol],     um1 = u_pad[pcol + 1], u0 = u_pad[pcol + 2];
                float up1 = u_pad[pcol + 3], up2 = u_pad[pcol + 4];
                float o1 = cf[0]
                    + cf[1]*um2 + cf[2]*um1 + cf[3]*u0 + cf[4]*up1 + cf[5]*up2
                    + cf[6]*um2*um2 + cf[7]*um1*um1 + cf[8]*u0*u0
                    + cf[9]*up1*up1 + cf[10]*up2*up2
                    + cf[11]*um2*um1 + cf[12]*um1*u0 + cf[13]*u0*up1
                    + cf[14]*up1*up2 + cf[15]*um2*u0 + cf[16]*um1*up1 + cf[17]*u0*up2;
                out[n * XD + pcol] = o1 + partial + other + b3v;
            }
        }
    }
}

extern "C" void kernel_launch(void* const* d_in, const int* in_sizes, int n_in,
                              void* d_out, int out_size, void* d_ws, size_t ws_size,
                              hipStream_t stream) {
    (void)in_sizes; (void)n_in; (void)out_size; (void)d_ws; (void)ws_size;
    // setup_inputs order: t, u, coeff, W1, b1, W2, b2, W3, b3
    const float* u     = (const float*)d_in[1];
    const float* coeff = (const float*)d_in[2];
    const float* W1    = (const float*)d_in[3];
    const float* b1    = (const float*)d_in[4];
    const float* W2    = (const float*)d_in[5];
    const float* b2    = (const float*)d_in[6];
    const float* W3    = (const float*)d_in[7];
    const float* b3    = (const float*)d_in[8];
    float* out = (float*)d_out;

    dim3 grid(1024), block(BLOCK_THREADS);   // 8192 waves -> 8 items/wave
    lorenz96_corr_kernel<<<grid, block, 0, stream>>>(u, coeff, W1, b1, W2, b2, W3, b3, out);
}
